// DecisionTransformer_84713934946715
// MI455X (gfx1250) — matrix operand order repaired and checked by reading, not compiled, not run on hardware
//
#include <hip/hip_runtime.h>
#include <hip/hip_bf16.h>
#include <math.h>

// ---------------- problem constants ----------------
#define B_   8
#define T_   256
#define S_   768
#define H_   1024
#define NH_  16
#define DH_  64
#define FF_  4096
#define L_   6
#define BS_  (B_*S_)     // 6144 tokens

typedef __attribute__((ext_vector_type(16))) __bf16 v16bf;
typedef __attribute__((ext_vector_type(8)))  __bf16 v8bf;
typedef __attribute__((ext_vector_type(8)))  float  v8f;
typedef int v4i __attribute__((vector_size(16)));

// ---- CDNA5 async global->LDS copy (ASYNCcnt path), guarded ------------------
// clang signature: (v4i addrspace(1)*, v4i addrspace(3)*, imm offset, imm cpol)
#if __has_builtin(__builtin_amdgcn_global_load_async_to_lds_b128)
#define HAVE_ASYNC_LDS 1
#define ASYNC_COPY_B128(gsrc, ldst)                                            \
    __builtin_amdgcn_global_load_async_to_lds_b128(                            \
        (__attribute__((address_space(1))) v4i*)(gsrc),                        \
        (__attribute__((address_space(3))) v4i*)(ldst), 0, 0)
#endif

#if __has_builtin(__builtin_amdgcn_s_wait_asynccnt)
#define WAIT_ASYNC0() __builtin_amdgcn_s_wait_asynccnt(0)
#else
#define WAIT_ASYNC0() asm volatile("s_wait_asynccnt 0" ::: "memory")
#endif

__device__ __forceinline__ v8f zero_v8f() {
    v8f z;
#pragma unroll
    for (int i = 0; i < 8; i++) z[i] = 0.0f;
    return z;
}

__device__ __forceinline__ v8bf pack8(const float* f) {
    union { v8bf v; __bf16 e[8]; } u;
#pragma unroll
    for (int t = 0; t < 8; t++) u.e[t] = (__bf16)f[t];
    return u.v;
}

__device__ __forceinline__ v8f wmma_bf16(v16bf a, v16bf b, v8f c) {
    // v_wmma_f32_16x16x32_bf16  D = A*B + C
    return __builtin_amdgcn_wmma_f32_16x16x32_bf16(
        /*neg_a=*/false, a, /*neg_b=*/false, b,
        /*c_mod=*/(short)0, c, /*reuse_a=*/false, /*reuse_b=*/false);
}

__device__ __forceinline__ float gelu_exact(float x) {
    return 0.5f * x * (1.0f + erff(x * 0.70710678118654752f));
}

// ---- fragment loaders ------------------------------------------------------
// A-matrix 16x32 bf16 ISA layout: lanes 0-15 (M=lane):  elems0-7 = K0-7,  elems8-15 = K16-23
//                                 lanes16-31 (M=lane-16): elems0-7 = K8-15, elems8-15 = K24-31
// LDS tile row-major [16][ld] bf16.
__device__ __forceinline__ v16bf frag_a_lds(const __bf16* base, int ld, int koff, int lane) {
    int c = lane & 15, h = lane >> 4;
    const __bf16* p = base + c * ld + koff;
    union { v16bf v; v8bf q[2]; } u;
    u.q[0] = *(const v8bf*)(p + h * 8);
    u.q[1] = *(const v8bf*)(p + 16 + h * 8);
    return u.v;
}
// B-matrix 32x16 bf16 layout: lanes 0-15 (N=lane): elems0-7 = K0-7,  elems8-15 = K16-23;
//                             lanes16-31 (N=lane-16): elems0-7 = K8-15, elems8-15 = K24-31
// LDS tile stored K-contiguous per output column (row n holds its K values, stride ld).
__device__ __forceinline__ v16bf frag_b_lds(const __bf16* base, int ld, int koff, int lane) {
    int c = lane & 15, h = lane >> 4;
    const __bf16* p = base + c * ld + koff;
    union { v16bf v; v8bf q[2]; } u;
    u.q[0] = *(const v8bf*)(p + h * 8);
    u.q[1] = *(const v8bf*)(p + 16 + h * 8);
    return u.v;
}
// A-fragment straight from fp32 global (rows stride ld), K window [koff, koff+32)
__device__ __forceinline__ v16bf frag_a_f32(const float* base, int ld, int koff,
                                            float scale, int lane) {
    int c = lane & 15, h = lane >> 4;
    const float* p = base + (size_t)c * ld + koff;
    union { v16bf v; __bf16 e[16]; } u;
#pragma unroll
    for (int t = 0; t < 8; t++) u.e[t]     = (__bf16)(p[h * 8 + t] * scale);
#pragma unroll
    for (int t = 0; t < 8; t++) u.e[t + 8] = (__bf16)(p[16 + h * 8 + t] * scale);
    return u.v;
}
// B-fragment (K^T style): column n of B is memory row n (stride ld); K dim contiguous
__device__ __forceinline__ v16bf frag_b_f32(const float* base, int ld, int koff, int lane) {
    int c = lane & 15, h = lane >> 4;
    const float* p = base + (size_t)c * ld + koff;
    union { v16bf v; __bf16 e[16]; } u;
#pragma unroll
    for (int t = 0; t < 8; t++) u.e[t] = (__bf16)p[h * 8 + t];
#pragma unroll
    for (int t = 0; t < 8; t++) u.e[t + 8] = (__bf16)p[16 + h * 8 + t];
    return u.v;
}

// ---------------- weight transpose: W[K][N] f32 -> WT[N][K] bf16 -------------
// tile 32x32 through LDS; grid (K/32, N/32); block 256. K,N multiples of 32.
__global__ void transpose_w_kernel(const float* __restrict__ W, __bf16* __restrict__ WT,
                                   int K, int N) {
    __shared__ float lt[32][33];
    const int k0 = blockIdx.x * 32;
    const int n0 = blockIdx.y * 32;
    const int c  = threadIdx.x & 31;
    const int r8 = threadIdx.x >> 5;           // 0..7
#pragma unroll
    for (int rr = r8; rr < 32; rr += 8)
        lt[rr][c] = W[(size_t)(k0 + rr) * N + n0 + c];
    __syncthreads();
#pragma unroll
    for (int rr = r8; rr < 32; rr += 8)
        WT[(size_t)(n0 + rr) * K + k0 + c] = (__bf16)lt[c][rr];
}

// ---------------- GEMM:  C[M,N] = op(A[M,K] @ W[K,N] + bias) (+res) ---------
// WT is the bf16 pre-transposed weight [N][K].
// block = 256 thr = 8 waves (2 M x 4 N); block tile 64x128; K step 64.
__global__ void gemm_bf16_kernel(const float* __restrict__ A,
                                 const __bf16* __restrict__ WT,
                                 const float* __restrict__ bias,
                                 const float* __restrict__ res,
                                 float* __restrict__ C,
                                 int M, int K, int N,
                                 int chunked, int rowOff, int doGelu) {
    __shared__ __align__(32) __bf16 lA[64 * 64];    //  8 KB, [m][k]
    __shared__ __align__(32) __bf16 lB[128 * 64];   // 16 KB, [n][k]

    const int tid  = threadIdx.x;
    const int wid  = tid >> 5;
    const int lane = tid & 31;
    const int wm   = wid >> 2;      // 0..1
    const int wn   = wid & 3;       // 0..3
    const int m0   = blockIdx.y * 64;
    const int n0   = blockIdx.x * 128;

    v8f acc[2][2];
#pragma unroll
    for (int i = 0; i < 2; i++)
#pragma unroll
        for (int j = 0; j < 2; j++) acc[i][j] = zero_v8f();

    // per-thread staging coordinates (uniform across K loop)
    const int ar_row = m0 + (tid >> 2);
    const int a_seg  = (tid & 3) * 16;
    const int a_src  = chunked ? ((ar_row >> 8) * S_ + rowOff + (ar_row & 255)) : ar_row;
    const int b_n    = tid >> 1;          // 0..127
    const int b_ks   = (tid & 1) * 32;    // 0 or 32
    const bool b_in  = (n0 + b_n < N);    // wave-uniform for our N sizes

    for (int k0 = 0; k0 < K; k0 += 64) {
        __syncthreads();   // previous iteration's consumers done

        // ---- stage B first: 128 n x 64 k bf16 copy [n][k] -> [n][k] ----
        // (async DMA to LDS when available; overlaps with the A-stage below)
        {
            __bf16* q = &lB[b_n * 64 + b_ks];
            if (b_in) {
                const __bf16* p = WT + (size_t)(n0 + b_n) * K + k0 + b_ks;
#ifdef HAVE_ASYNC_LDS
#pragma unroll
                for (int t = 0; t < 4; t++)
                    ASYNC_COPY_B128(p + t * 8, q + t * 8);
#else
                *(v8bf*)(q)      = *(const v8bf*)(p);
                *(v8bf*)(q + 8)  = *(const v8bf*)(p + 8);
                *(v8bf*)(q + 16) = *(const v8bf*)(p + 16);
                *(v8bf*)(q + 24) = *(const v8bf*)(p + 24);
#endif
                if (k0 + 64 < K) __builtin_prefetch(p + 64, 0, 1);
            } else {
                float zbuf[8] = {0, 0, 0, 0, 0, 0, 0, 0};
                *(v8bf*)(q)      = pack8(zbuf);
                *(v8bf*)(q + 8)  = pack8(zbuf);
                *(v8bf*)(q + 16) = pack8(zbuf);
                *(v8bf*)(q + 24) = pack8(zbuf);
            }
        }
        // ---- stage A: 64 rows x 64 K (fp32 -> bf16), vector loads/stores ----
        {
            const float* p = A + (size_t)a_src * K + k0 + a_seg;
            float buf[16];
#pragma unroll
            for (int t = 0; t < 4; t++) {
                float4 f = ((const float4*)p)[t];
                buf[4 * t] = f.x; buf[4 * t + 1] = f.y;
                buf[4 * t + 2] = f.z; buf[4 * t + 3] = f.w;
            }
            __bf16* q = &lA[(tid >> 2) * 64 + a_seg];
            *(v8bf*)q       = pack8(buf);
            *(v8bf*)(q + 8) = pack8(buf + 8);
            if (k0 + 64 < K) __builtin_prefetch(p + 64, 0, 1);
        }
#ifdef HAVE_ASYNC_LDS
        WAIT_ASYNC0();     // this wave's async B copies have landed in LDS
#endif
        __syncthreads();

        // ---- compute: 2 K-chunks x (2 M x 2 N) WMMAs, all-vector LDS reads ----
#pragma unroll
        for (int kc = 0; kc < 64; kc += 32) {
            v16bf b0 = frag_b_lds(&lB[(wn * 32 + 0)  * 64], 64, kc, lane);
            v16bf b1 = frag_b_lds(&lB[(wn * 32 + 16) * 64], 64, kc, lane);
#pragma unroll
            for (int i = 0; i < 2; i++) {
                v16bf af = frag_a_lds(&lA[(wm * 32 + i * 16) * 64], 64, kc, lane);
                acc[i][0] = wmma_bf16(af, b0, acc[i][0]);
                acc[i][1] = wmma_bf16(af, b1, acc[i][1]);
            }
        }
    }

    // epilogue: D layout -> row = r + 8*(lane/16), col = lane%16
    const int c = lane & 15, hf = lane >> 4;
#pragma unroll
    for (int i = 0; i < 2; i++)
#pragma unroll
        for (int j = 0; j < 2; j++) {
            int nn = n0 + wn * 32 + j * 16 + c;
            if (nn < N) {
                float bv = bias[nn];
#pragma unroll
                for (int r = 0; r < 8; r++) {
                    int mm = m0 + wm * 32 + i * 16 + hf * 8 + r;
                    float x = acc[i][j][r] + bv;
                    if (doGelu) x = gelu_exact(x);
                    if (res)    x += res[(size_t)mm * N + nn];
                    C[(size_t)mm * N + nn] = x;
                }
            }
        }
}

// ---------------- causal flash attention ------------------------------------
// grid (S/128, B*NH), block 256 = 8 waves; each wave owns a 16-query tile.
__global__ void attn_kernel(const float* __restrict__ q,
                            const float* __restrict__ k,
                            const float* __restrict__ v,
                            float* __restrict__ o) {
    __shared__ __align__(32) __bf16 lP[8][16 * 32];   //  8 KB
    __shared__ __align__(32) __bf16 lV[8][64 * 32];   // 32 KB (d-major, key contig)

    const int tid  = threadIdx.x;
    const int wid  = tid >> 5;
    const int lane = tid & 31;
    const int c    = lane & 15, hf = lane >> 4;
    const int qb   = (blockIdx.x * 8 + wid) * 16;
    const int b    = blockIdx.y / NH_;
    const int hd   = blockIdx.y % NH_;

    const size_t boff = (size_t)b * S_ * H_ + (size_t)hd * DH_;
    const float* qp = q + boff;
    const float* kp = k + boff;
    const float* vp = v + boff;
    float*       op = o + boff;

    // Q fragments (scaled by 1/sqrt(64))
    v16bf qf0 = frag_a_f32(qp + (size_t)qb * H_, H_, 0,  0.125f, lane);
    v16bf qf1 = frag_a_f32(qp + (size_t)qb * H_, H_, 32, 0.125f, lane);

    float row_m[8], row_l[8];
#pragma unroll
    for (int r = 0; r < 8; r++) { row_m[r] = -3.0e38f; row_l[r] = 0.0f; }
    v8f oacc[4];
#pragma unroll
    for (int j = 0; j < 4; j++) oacc[j] = zero_v8f();

    for (int kb = 0; kb <= qb + 15; kb += 32) {
        // prefetch next key block's K/V rows (one row per lane)
        if (kb + 32 <= qb + 15) {
            __builtin_prefetch(kp + (size_t)(kb + 32 + lane) * H_, 0, 1);
            __builtin_prefetch(vp + (size_t)(kb + 32 + lane) * H_, 0, 1);
        }
        // stage this key block's V (32 keys x 64 d) transposed into per-wave LDS
        {
            const float* vrow = vp + (size_t)(kb + lane) * H_;
            __bf16* lv = &lV[wid][0];
#pragma unroll
            for (int d = 0; d < 64; d++) lv[d * 32 + lane] = (__bf16)vrow[d];
        }

        // scores: two 16x16 tiles (keys kb..kb+15, kb+16..kb+31)
        v8f s0 = zero_v8f(), s1 = zero_v8f();
        s0 = wmma_bf16(qf0, frag_b_f32(kp + (size_t)kb * H_, H_, 0,  lane), s0);
        s0 = wmma_bf16(qf1, frag_b_f32(kp + (size_t)kb * H_, H_, 32, lane), s0);
        s1 = wmma_bf16(qf0, frag_b_f32(kp + (size_t)(kb + 16) * H_, H_, 0,  lane), s1);
        s1 = wmma_bf16(qf1, frag_b_f32(kp + (size_t)(kb + 16) * H_, H_, 32, lane), s1);

        // causal mask
#pragma unroll
        for (int r = 0; r < 8; r++) {
            int rowg = qb + hf * 8 + r;
            if (kb + c      > rowg) s0[r] = -3.0e38f;
            if (kb + 16 + c > rowg) s1[r] = -3.0e38f;
        }
        // row max across 16 lanes of each half-wave
        float tmax[8];
#pragma unroll
        for (int r = 0; r < 8; r++) tmax[r] = fmaxf(s0[r], s1[r]);
#pragma unroll
        for (int off = 8; off; off >>= 1)
#pragma unroll
            for (int r = 0; r < 8; r++)
                tmax[r] = fmaxf(tmax[r], __shfl_xor(tmax[r], off, 16));

        float sc[8];
#pragma unroll
        for (int r = 0; r < 8; r++) {
            float mn = fmaxf(row_m[r], tmax[r]);
            sc[r] = __expf(row_m[r] - mn);
            row_m[r] = mn;
        }
#pragma unroll
        for (int r = 0; r < 8; r++) {
            s0[r] = __expf(s0[r] - row_m[r]);
            s1[r] = __expf(s1[r] - row_m[r]);
        }
        float rsum[8];
#pragma unroll
        for (int r = 0; r < 8; r++) rsum[r] = s0[r] + s1[r];
#pragma unroll
        for (int off = 8; off; off >>= 1)
#pragma unroll
            for (int r = 0; r < 8; r++)
                rsum[r] += __shfl_xor(rsum[r], off, 16);
#pragma unroll
        for (int r = 0; r < 8; r++) row_l[r] = row_l[r] * sc[r] + rsum[r];
#pragma unroll
        for (int j = 0; j < 4; j++)
#pragma unroll
            for (int r = 0; r < 8; r++) oacc[j][r] *= sc[r];

        // P (D-layout) -> LDS -> A-fragment re-layout
        {
            __bf16* pp = &lP[wid][0];
#pragma unroll
            for (int r = 0; r < 8; r++) {
                pp[(hf * 8 + r) * 32 + c]      = (__bf16)s0[r];
                pp[(hf * 8 + r) * 32 + 16 + c] = (__bf16)s1[r];
            }
        }
        asm volatile("s_wait_dscnt 0" ::: "memory");  // cross-lane LDS RAW
        v16bf pf = frag_a_lds(&lP[wid][0], 32, 0, lane);
#pragma unroll
        for (int j = 0; j < 4; j++) {
            v16bf vf = frag_b_lds(&lV[wid][(j * 16) * 32], 32, 0, lane);
            oacc[j] = wmma_bf16(pf, vf, oacc[j]);
        }
    }

    // write O = acc / l
#pragma unroll
    for (int j = 0; j < 4; j++)
#pragma unroll
        for (int r = 0; r < 8; r++) {
            int rowg = qb + hf * 8 + r;
            op[(size_t)rowg * H_ + j * 16 + c] = oacc[j][r] / row_l[r];
        }
}

// ---------------- embedding + interleave ------------------------------------
__global__ void embed_kernel(const int* __restrict__ ts, const int* __restrict__ st,
                             const int* __restrict__ ac, const float* __restrict__ rtg,
                             const float* __restrict__ Wt, const float* __restrict__ Ws,
                             const float* __restrict__ Wa, const float* __restrict__ Wr,
                             const float* __restrict__ br, float* __restrict__ out) {
    int row = blockIdx.x;              // 0..BS-1
    int b = row / S_, s = row % S_;
    int t = s / 3,  j = s % 3;
    const float* te = Wt + (size_t)ts[b * T_ + t] * H_;
    float* orow = out + (size_t)row * H_;
    if (j == 0) {
        float rv = rtg[b * T_ + t];
        for (int i = threadIdx.x; i < H_; i += blockDim.x)
            orow[i] = rv * Wr[i] + br[i] + te[i];
    } else if (j == 1) {
        const float* se = Ws + (size_t)st[b * T_ + t] * H_;
        for (int i = threadIdx.x; i < H_; i += blockDim.x) orow[i] = se[i] + te[i];
    } else {
        const float* ae = Wa + (size_t)ac[b * T_ + t] * H_;
        for (int i = threadIdx.x; i < H_; i += blockDim.x) orow[i] = ae[i] + te[i];
    }
}

// ---------------- LayerNorm (one block per row) ------------------------------
__global__ void ln_kernel(const float* __restrict__ x, const float* __restrict__ g,
                          const float* __restrict__ bta, float* __restrict__ out) {
    __shared__ float rs[256], rs2[256];
    int row = blockIdx.x, tid = threadIdx.x;
    const float* xr = x + (size_t)row * H_;
    float s = 0.f, s2 = 0.f;
    for (int i = tid; i < H_; i += 256) { float v = xr[i]; s += v; s2 += v * v; }
    rs[tid] = s; rs2[tid] = s2;
    __syncthreads();
    for (int st = 128; st > 0; st >>= 1) {
        if (tid < st) { rs[tid] += rs[tid + st]; rs2[tid] += rs2[tid + st]; }
        __syncthreads();
    }
    float mean = rs[0] * (1.0f / H_);
    float var  = rs2[0] * (1.0f / H_) - mean * mean;
    float rstd = rsqrtf(var + 1e-5f);
    for (int i = tid; i < H_; i += 256)
        out[(size_t)row * H_ + i] = (xr[i] - mean) * rstd * g[i] + bta[i];
}

// ---------------- rtg head (N=1 dot per row) ---------------------------------
__global__ void rtg_head_kernel(const float* __restrict__ h, const float* __restrict__ Wr,
                                const float* __restrict__ br, float* __restrict__ out) {
    int wid = threadIdx.x >> 5, lane = threadIdx.x & 31;
    int m = blockIdx.x * 8 + wid;            // 0..2047
    int b = m >> 8, t = m & 255;
    const float* row = h + (size_t)(b * S_ + 512 + t) * H_;
    float acc = 0.f;
    for (int i = lane; i < H_; i += 32) acc += row[i] * Wr[i];
#pragma unroll
    for (int off = 16; off; off >>= 1) acc += __shfl_xor(acc, off, 32);
    if (lane == 0) out[m] = acc + br[0];
}

// ---------------- driver -----------------------------------------------------
extern "C" void kernel_launch(void* const* d_in, const int* in_sizes, int n_in,
                              void* d_out, int out_size, void* d_ws, size_t ws_size,
                              hipStream_t stream) {
    const int*   timesteps = (const int*)d_in[0];
    const int*   states    = (const int*)d_in[1];
    const int*   actions   = (const int*)d_in[2];
    const float* rtg       = (const float*)d_in[3];
    const float* Wt  = (const float*)d_in[4];
    const float* Ws  = (const float*)d_in[5];
    const float* Wa  = (const float*)d_in[6];
    const float* Wr  = (const float*)d_in[7];
    const float* br  = (const float*)d_in[8];
    const float* lng = (const float*)d_in[9];
    const float* lnb = (const float*)d_in[10];
    const float* Wq  = (const float*)d_in[11];
    const float* bq  = (const float*)d_in[12];
    const float* Wk  = (const float*)d_in[13];
    const float* bk  = (const float*)d_in[14];
    const float* Wv  = (const float*)d_in[15];
    const float* bv  = (const float*)d_in[16];
    const float* Wp  = (const float*)d_in[17];
    const float* bp  = (const float*)d_in[18];
    const float* W1  = (const float*)d_in[19];
    const float* b1  = (const float*)d_in[20];
    const float* W2  = (const float*)d_in[21];
    const float* b2  = (const float*)d_in[22];
    const float* ln1g = (const float*)d_in[23];
    const float* ln1b = (const float*)d_in[24];
    const float* ln2g = (const float*)d_in[25];
    const float* ln2b = (const float*)d_in[26];
    const float* Wrtg = (const float*)d_in[27];
    const float* brtg = (const float*)d_in[28];
    const float* Wst  = (const float*)d_in[29];
    const float* bst  = (const float*)d_in[30];
    const float* Wac  = (const float*)d_in[31];
    const float* bac  = (const float*)d_in[32];

    float* ws = (float*)d_ws;
    float* hA = ws;                                 // BS x H
    float* tb = hA + (size_t)BS_ * H_;              // BS x H
    float* R  = tb + (size_t)BS_ * H_;              // BS x FF (reused)
    float* qb_ = R;
    float* kb_ = R + (size_t)BS_ * H_;
    float* vb_ = R + (size_t)2 * BS_ * H_;
    float* ob_ = R + (size_t)3 * BS_ * H_;
    float* mid = R;
    __bf16* WT = (__bf16*)(R + (size_t)BS_ * FF_);  // rotating [N][K] bf16, 8 MB max

    float* out_state  = (float*)d_out;                          // 2048 x 1024
    float* out_action = out_state + (size_t)2048 * 1024;        // 2048 x 64
    float* out_rtg    = out_action + (size_t)2048 * 64;         // 2048 x 1

    dim3 blk(256);
    // transpose W[K][N] f32 -> WT[N][K] bf16, then WMMA GEMM against WT
    auto gemmT = [&](const float* A, const float* W, const float* bias, const float* res,
                     float* C, int M, int K, int N, int chunked, int rowOff, int gelu) {
        hipLaunchKernelGGL(transpose_w_kernel, dim3(K / 32, N / 32), blk, 0, stream,
                           W, WT, K, N);
        dim3 g((N + 127) / 128, M / 64);
        hipLaunchKernelGGL(gemm_bf16_kernel, g, blk, 0, stream,
                           A, WT, bias, res, C, M, K, N, chunked, rowOff, gelu);
    };

    // embedding + first LN
    hipLaunchKernelGGL(embed_kernel, dim3(BS_), blk, 0, stream,
                       timesteps, states, actions, rtg, Wt, Ws, Wa, Wr, br, tb);
    hipLaunchKernelGGL(ln_kernel, dim3(BS_), blk, 0, stream, tb, lng, lnb, hA);

    for (int l = 0; l < L_; l++) {
        const float* Wql = Wq + (size_t)l * H_ * H_;
        const float* Wkl = Wk + (size_t)l * H_ * H_;
        const float* Wvl = Wv + (size_t)l * H_ * H_;
        const float* Wpl = Wp + (size_t)l * H_ * H_;
        const float* W1l = W1 + (size_t)l * H_ * FF_;
        const float* W2l = W2 + (size_t)l * FF_ * H_;

        gemmT(hA, Wql, bq + l * H_, nullptr, qb_, BS_, H_, H_, 0, 0, 0);
        gemmT(hA, Wkl, bk + l * H_, nullptr, kb_, BS_, H_, H_, 0, 0, 0);
        gemmT(hA, Wvl, bv + l * H_, nullptr, vb_, BS_, H_, H_, 0, 0, 0);

        hipLaunchKernelGGL(attn_kernel, dim3(S_ / 128, B_ * NH_), blk, 0, stream,
                           qb_, kb_, vb_, ob_);

        gemmT(ob_, Wpl, bp + l * H_, hA, tb, BS_, H_, H_, 0, 0, 0);          // + residual
        hipLaunchKernelGGL(ln_kernel, dim3(BS_), blk, 0, stream,
                           tb, ln1g + l * H_, ln1b + l * H_, hA);

        gemmT(hA, W1l, b1 + l * FF_, nullptr, mid, BS_, H_, FF_, 0, 0, 1);   // GELU
        gemmT(mid, W2l, b2 + l * H_, hA, tb, BS_, FF_, H_, 0, 0, 0);         // + residual
        hipLaunchKernelGGL(ln_kernel, dim3(BS_), blk, 0, stream,
                           tb, ln2g + l * H_, ln2b + l * H_, hA);
    }

    // heads: h.reshape(B,3,T,H) -> chunk1 rows at +256, chunk2 rows at +512
    gemmT(hA, Wst, bst, nullptr, out_state,  B_ * T_, H_, 1024, 1, 512, 0);
    gemmT(hA, Wac, bac, nullptr, out_action, B_ * T_, H_, 64,   1, 256, 0);
    hipLaunchKernelGGL(rtg_head_kernel, dim3(256), blk, 0, stream,
                       hA, Wrtg, brtg, out_rtg);
}
